// DiffTransformerModel_79903571575225
// MI455X (gfx1250) — compile-verified, matrix operand order checked by reading, not run on hardware
//
#include <hip/hip_runtime.h>

typedef __bf16 v16bf __attribute__((ext_vector_type(16)));
typedef float  v8f   __attribute__((ext_vector_type(8)));

union AFrag { uint4 u[2]; v16bf v; };

static __device__ __forceinline__ v8f wmma_bf16(v16bf a, v16bf b, v8f c) {
    // D = A(16x32 bf16) x B(32x16 bf16) + C(16x16 f32)
    return __builtin_amdgcn_wmma_f32_16x16x32_bf16(false, a, false, b, (short)0, c,
                                                   false, false);
}

static __device__ __forceinline__ v8f zero8() {
    v8f z = {0.f, 0.f, 0.f, 0.f, 0.f, 0.f, 0.f, 0.f};
    return z;
}

// ---------------------------------------------------------------------------
// fp32 -> bf16 convert
// ---------------------------------------------------------------------------
__global__ __launch_bounds__(256) void cvt_f32_bf16(const float* __restrict__ in,
                                                    __bf16* __restrict__ out, int n) {
    int i = blockIdx.x * 256 + threadIdx.x;
    if (i < n) out[i] = (__bf16)in[i];
}

// ---------------------------------------------------------------------------
// fp32 [L][K][N] -> bf16 [L][N][K] transpose-convert (for Wq/Wk/Wv)
// ---------------------------------------------------------------------------
__global__ __launch_bounds__(256) void transpose_cvt(const float* __restrict__ in,
                                                     __bf16* __restrict__ out,
                                                     int K, int N) {
    int idx = blockIdx.x * 256 + threadIdx.x;           // over N*K
    if (idx >= K * N) return;
    size_t base = (size_t)blockIdx.y * (size_t)K * (size_t)N;
    int n = idx / K, k = idx % K;
    out[base + idx] = (__bf16)in[base + (size_t)k * N + n];
}

// ---------------------------------------------------------------------------
// C[M][N] = A[M][K](bf16) x Wt[N][K](bf16)^T + bias (+PE)
// Each wave: one 16-row x 64-col strip (4 N-tiles, shared A frag).
// Block = 8 waves covers 16 x 512. Grid = (M/16, N/512).
// STORE_MODE: 0 = bf16 row-major, 1 = bf16 transposed [B][N][S] (for V), 2 = f32.
// ---------------------------------------------------------------------------
template <int STORE_MODE, bool ADD_PE>
__global__ __launch_bounds__(256) void gemm_wmma(const __bf16* __restrict__ A,
                                                 const __bf16* __restrict__ Wt,
                                                 const float* __restrict__ bias,
                                                 void* __restrict__ outp,
                                                 int M, int N, int Kdim, int SperB) {
    int lane = threadIdx.x & 31;
    int wv   = threadIdx.x >> 5;
    int m0   = blockIdx.x * 16;
    int n0   = blockIdx.y * 512 + wv * 64;
    if (n0 >= N) return;

    int row   = lane & 15;            // A-row / B-col / C-col index
    int khalf = (lane >> 4) << 3;     // 0 or 8 : K-half split across lane halves

    const __bf16* arow = A + (size_t)(m0 + row) * Kdim;
    const __bf16* brows[4];
#pragma unroll
    for (int t = 0; t < 4; ++t)
        brows[t] = Wt + (size_t)(n0 + t * 16 + row) * Kdim;

    v8f acc[4];
#pragma unroll
    for (int t = 0; t < 4; ++t) acc[t] = zero8();

    for (int k0 = 0; k0 < Kdim; k0 += 32) {
        // stage all fragments first (one clause of b128 loads, one wait) ...
        AFrag af, bfr[4];
        af.u[0] = *(const uint4*)(arow + k0 + khalf);
        af.u[1] = *(const uint4*)(arow + k0 + 16 + khalf);
#pragma unroll
        for (int t = 0; t < 4; ++t) {
            bfr[t].u[0] = *(const uint4*)(brows[t] + k0 + khalf);
            bfr[t].u[1] = *(const uint4*)(brows[t] + k0 + 16 + khalf);
        }
        __builtin_prefetch(arow + k0 + 128, 0, 1);
        // ... then a back-to-back WMMA burst (distinct accumulators: no hazards)
#pragma unroll
        for (int t = 0; t < 4; ++t)
            acc[t] = wmma_bf16(af.v, bfr[t].v, acc[t]);
    }

    int rbase = khalf;                // C row base: 0 or 8
#pragma unroll
    for (int t = 0; t < 4; ++t) {
        int   gn = n0 + t * 16 + row;
        float bv = bias ? bias[gn] : 0.f;
        float vals[8];
#pragma unroll
        for (int j = 0; j < 8; ++j) {
            int   gm = m0 + rbase + j;
            float v  = acc[t][j] + bv;
            if (ADD_PE) {
                float pos  = (float)(gm % SperB);
                float freq = __expf(-9.210340371976184f * (float)(gn & ~1) / (float)N);
                float ang  = pos * freq;
                v += (gn & 1) ? __cosf(ang) : __sinf(ang);
            }
            vals[j] = v;
        }
        if (STORE_MODE == 2) {
            float* op = (float*)outp;
#pragma unroll
            for (int j = 0; j < 8; ++j)
                op[(size_t)(m0 + rbase + j) * N + gn] = vals[j];
        } else if (STORE_MODE == 0) {
            __bf16* op = (__bf16*)outp;
#pragma unroll
            for (int j = 0; j < 8; ++j)
                op[(size_t)(m0 + rbase + j) * N + gn] = (__bf16)vals[j];
        } else {
            // transposed: out[(b*N + gn)*SperB + s]; 8 consecutive rows -> 1x b128
            __bf16* op  = (__bf16*)outp;
            int     gm0 = m0 + rbase;
            int     b   = gm0 / SperB, s0 = gm0 % SperB;
            __align__(16) __bf16 tmp[8];
#pragma unroll
            for (int j = 0; j < 8; ++j) tmp[j] = (__bf16)vals[j];
            *(uint4*)(op + ((size_t)(b * N + gn)) * SperB + s0) = *(const uint4*)tmp;
        }
    }
}

// ---------------------------------------------------------------------------
// Differential flash attention.
//   scores = softmax(Q1K1^T * s) - 0.5 * softmax(Q2K2^T * s);  O = scores @ V
// Grid: (S/16, B), block 256 (8 waves). Each block: 16 queries.
// Waves 0-3 compute S1 logits (dims 0..511, 128 dims each), waves 4-7 S2;
// partial 16x64 logit tiles reduced into LDS via ds_add_f32.
// Wave 0 performs the two online-softmax updates; every wave keeps
// O1/O2 f32 accumulators for its 128 output columns and does P@V WMMAs.
// V is pre-transposed [B][D][S] so B-fragments are contiguous b128 loads.
// ---------------------------------------------------------------------------
__global__ __launch_bounds__(256) void diff_attn(const __bf16* __restrict__ Q,
                                                 const __bf16* __restrict__ Km,
                                                 const __bf16* __restrict__ Vt,
                                                 __bf16* __restrict__ Out) {
    const int   S = 2048, D = 1024, HALF = 512, KT = 64;
    const float SCALE = 0.03125f;  // 1/sqrt(1024)

    int lane = threadIdx.x & 31;
    int wv   = threadIdx.x >> 5;
    int b    = blockIdx.y;
    int q0   = blockIdx.x * 16;

    __shared__ __align__(16) float  Sc[2][16][KT];   // raw logits (chunk)
    __shared__ __align__(16) __bf16 Pm[2][16][KT];   // exp(logit - m) as bf16
    __shared__ float stat_m[2][16], stat_l[2][16], stat_s[2][16];

    int row   = lane & 15;
    int khalf = (lane >> 4) << 3;
    int rbase = khalf;

    if (threadIdx.x < 32) {
        int mt = threadIdx.x >> 4, r = threadIdx.x & 15;
        stat_m[mt][r] = -1e30f;
        stat_l[mt][r] = 0.f;
    }

    // This wave's logit role: 128-dim slice of Q1 (waves 0-3) or Q2 (4-7).
    int           mat  = wv >> 2;
    int           dofs = mat * HALF + (wv & 3) * 128;
    const __bf16* qrow = Q + ((size_t)(b * S + q0 + row)) * D;
    AFrag         qf[4];
#pragma unroll
    for (int ks = 0; ks < 4; ++ks) {
        qf[ks].u[0] = *(const uint4*)(qrow + dofs + ks * 32 + khalf);
        qf[ks].u[1] = *(const uint4*)(qrow + dofs + ks * 32 + 16 + khalf);
    }

    int c0 = wv * 128;  // this wave's output columns
    // per-column V row base pointers (hoisted address arithmetic)
    const __bf16* vrows[8];
#pragma unroll
    for (int t = 0; t < 8; ++t)
        vrows[t] = Vt + ((size_t)(b * D + c0 + t * 16 + row)) * S;

    v8f O1[8], O2[8];
#pragma unroll
    for (int t = 0; t < 8; ++t) { O1[t] = zero8(); O2[t] = zero8(); }

    __syncthreads();

    for (int kb = 0; kb < S; kb += KT) {
        // (a) zero logit chunk
        {
            float* p = &Sc[0][0][0];
            for (int i = threadIdx.x; i < 2 * 16 * KT; i += 256) p[i] = 0.f;
        }
        __syncthreads();

        // (b) partial logits: 4 key subtiles x 4 K-steps of 32 dims
#pragma unroll
        for (int t = 0; t < 4; ++t) {
            int           key  = kb + t * 16 + row;
            const __bf16* krow = Km + ((size_t)(b * S + key)) * D;
            AFrag         kf[4];
#pragma unroll
            for (int ks = 0; ks < 4; ++ks) {            // stage 8 b128 loads
                kf[ks].u[0] = *(const uint4*)(krow + dofs + ks * 32 + khalf);
                kf[ks].u[1] = *(const uint4*)(krow + dofs + ks * 32 + 16 + khalf);
            }
            v8f sacc = zero8();
#pragma unroll
            for (int ks = 0; ks < 4; ++ks)              // WMMA burst
                sacc = wmma_bf16(qf[ks].v, kf[ks].v, sacc);
#pragma unroll
            for (int j = 0; j < 8; ++j)
                atomicAdd(&Sc[mat][rbase + j][t * 16 + row], sacc[j] * SCALE);
        }
        __syncthreads();

        // (c) online softmax update for both matrices (wave 0)
        if (threadIdx.x < 32) {
            int   mt = threadIdx.x >> 4, r = threadIdx.x & 15;
            float mo = stat_m[mt][r];
            float mx = mo;
            for (int c = 0; c < KT; ++c) mx = fmaxf(mx, Sc[mt][r][c]);
            float scf = __expf(mo - mx);
            float sum = 0.f;
            for (int c = 0; c < KT; ++c) {
                float p      = __expf(Sc[mt][r][c] - mx);
                Pm[mt][r][c] = (__bf16)p;
                sum += p;
            }
            stat_m[mt][r] = mx;
            stat_l[mt][r] = stat_l[mt][r] * scf + sum;
            stat_s[mt][r] = scf;
        }
        __syncthreads();

        // (d) rescale accumulators, then P @ V
#pragma unroll
        for (int j = 0; j < 8; ++j) {
            float f1 = stat_s[0][rbase + j];
            float f2 = stat_s[1][rbase + j];
#pragma unroll
            for (int t = 0; t < 8; ++t) { O1[t][j] *= f1; O2[t][j] *= f2; }
        }
#pragma unroll
        for (int ks = 0; ks < 2; ++ks) {
            // stage P fragments (LDS b128) and all 8 V fragments (global b128):
            // one clause of loads, then 16 back-to-back WMMAs.
            AFrag p1f, p2f, vf[8];
            p1f.u[0] = *(const uint4*)(&Pm[0][row][ks * 32 + khalf]);
            p1f.u[1] = *(const uint4*)(&Pm[0][row][ks * 32 + 16 + khalf]);
            p2f.u[0] = *(const uint4*)(&Pm[1][row][ks * 32 + khalf]);
            p2f.u[1] = *(const uint4*)(&Pm[1][row][ks * 32 + 16 + khalf]);
#pragma unroll
            for (int t = 0; t < 8; ++t) {
                const __bf16* vrow = vrows[t] + kb + ks * 32;
                vf[t].u[0] = *(const uint4*)(vrow + khalf);
                vf[t].u[1] = *(const uint4*)(vrow + 16 + khalf);
            }
#pragma unroll
            for (int t = 0; t < 8; ++t) {
                O1[t] = wmma_bf16(p1f.v, vf[t].v, O1[t]);
                O2[t] = wmma_bf16(p2f.v, vf[t].v, O2[t]);
            }
        }
        __syncthreads();
    }

    // epilogue: O = O1/l1 - 0.5 * O2/l2
#pragma unroll
    for (int j = 0; j < 8; ++j) {
        float il1 = 1.0f / stat_l[0][rbase + j];
        float il2 = 0.5f / stat_l[1][rbase + j];
        int   gm  = b * S + q0 + rbase + j;
#pragma unroll
        for (int t = 0; t < 8; ++t) {
            float v                                 = O1[t][j] * il1 - O2[t][j] * il2;
            Out[(size_t)gm * D + c0 + t * 16 + row] = (__bf16)v;
        }
    }
}

// ---------------------------------------------------------------------------
// Host orchestration
// ---------------------------------------------------------------------------
extern "C" void kernel_launch(void* const* d_in, const int* in_sizes, int n_in,
                              void* d_out, int out_size, void* d_ws, size_t ws_size,
                              hipStream_t stream) {
    (void)in_sizes; (void)n_in; (void)out_size; (void)ws_size;

    const int B = 8, S = 2048, IN = 512, D = 1024, NL = 6, OUT = 512;
    const int M = B * S;  // 16384

    const float* x     = (const float*)d_in[0];
    const float* W_in  = (const float*)d_in[1];
    const float* b_in  = (const float*)d_in[2];
    const float* W_ctx = (const float*)d_in[3];
    const float* b_ctx = (const float*)d_in[4];
    const float* Wq    = (const float*)d_in[5];
    const float* Wk    = (const float*)d_in[6];
    const float* Wv    = (const float*)d_in[7];
    const float* W_out = (const float*)d_in[8];
    const float* b_out = (const float*)d_in[9];

    // workspace layout (bf16), ~200 MB total
    __bf16* ws    = (__bf16*)d_ws;
    size_t  HS    = (size_t)M * D;        // 16,777,216 elements
    __bf16* hA    = ws;
    __bf16* hB    = hA + HS;
    __bf16* Qb    = hB + HS;
    __bf16* Kb    = Qb + HS;
    __bf16* Vt    = Kb + HS;              // transposed V: [B][D][S]
    __bf16* WinT  = Vt + HS;              // [1024][512]  (already [N][K])
    __bf16* WctxT = WinT + (size_t)D * IN;
    __bf16* WqT   = WctxT + (size_t)D * D;
    __bf16* WkT   = WqT + (size_t)NL * D * D;
    __bf16* WvT   = WkT + (size_t)NL * D * D;
    __bf16* WoutT = WvT + (size_t)NL * D * D;  // [512][1024] (already [N][K])

    // --- weight / input conversion (bf16 weights stay resident in 192MB L2) ---
    {
        int n = M * IN;  // x -> bf16 (into hB, reused as first GEMM input)
        cvt_f32_bf16<<<dim3((n + 255) / 256), dim3(256), 0, stream>>>(x, hB, n);
        n = D * IN;
        cvt_f32_bf16<<<dim3((n + 255) / 256), dim3(256), 0, stream>>>(W_in, WinT, n);
        n = D * D;
        cvt_f32_bf16<<<dim3((n + 255) / 256), dim3(256), 0, stream>>>(W_ctx, WctxT, n);
        n = OUT * D;
        cvt_f32_bf16<<<dim3((n + 255) / 256), dim3(256), 0, stream>>>(W_out, WoutT, n);
        // Wq/Wk/Wv are used un-transposed (h @ W) -> store as [N][K]
        dim3 tg((D * D + 255) / 256, NL);
        transpose_cvt<<<tg, dim3(256), 0, stream>>>(Wq, WqT, D, D);
        transpose_cvt<<<tg, dim3(256), 0, stream>>>(Wk, WkT, D, D);
        transpose_cvt<<<tg, dim3(256), 0, stream>>>(Wv, WvT, D, D);
    }

    // --- h = x @ W_in^T + b_in ---
    gemm_wmma<0, false><<<dim3(M / 16, D / 512), dim3(256), 0, stream>>>(
        hB, WinT, b_in, hA, M, D, IN, S);
    // --- h = h @ W_ctx^T + b_ctx + PE ---
    gemm_wmma<0, true><<<dim3(M / 16, D / 512), dim3(256), 0, stream>>>(
        hA, WctxT, b_ctx, hB, M, D, D, S);

    __bf16* h  = hB;
    __bf16* hn = hA;
    for (int i = 0; i < NL; ++i) {
        size_t wofs = (size_t)i * D * D;
        gemm_wmma<0, false><<<dim3(M / 16, D / 512), dim3(256), 0, stream>>>(
            h, WqT + wofs, nullptr, Qb, M, D, D, S);
        gemm_wmma<0, false><<<dim3(M / 16, D / 512), dim3(256), 0, stream>>>(
            h, WkT + wofs, nullptr, Kb, M, D, D, S);
        gemm_wmma<1, false><<<dim3(M / 16, D / 512), dim3(256), 0, stream>>>(
            h, WvT + wofs, nullptr, Vt, M, D, D, S);
        diff_attn<<<dim3(S / 16, B), dim3(256), 0, stream>>>(Qb, Kb, Vt, hn);
        __bf16* t = h; h = hn; hn = t;
    }

    // --- out = h @ W_out^T + b_out (fp32) ---
    gemm_wmma<2, false><<<dim3(M / 16, OUT / 512), dim3(256), 0, stream>>>(
        h, WoutT, b_out, (float*)d_out, M, OUT, D, S);
}
